// HGNNConv_9216999817273
// MI455X (gfx1250) — matrix operand order repaired and checked by reading, not compiled, not run on hardware
//
#include <hip/hip_runtime.h>
#include <hip/hip_bf16.h>

typedef __attribute__((ext_vector_type(16))) _Float16 v16h;
typedef __attribute__((ext_vector_type(8)))  float    v8f;

// ---------------------------------------------------------------------------
// Device-scope HW fp32 atomic add (cross-WGP accumulation => DEV scope needed;
// inline asm guarantees global_atomic_add_f32 instead of a CAS loop).
// ---------------------------------------------------------------------------
__device__ __forceinline__ void gatomic_fadd(float* addr, float val) {
    asm volatile("global_atomic_add_f32 %0, %1, off scope:SCOPE_DEV"
                 :: "v"(addr), "v"(val) : "memory");
}

// ---------------------------------------------------------------------------
// Zero fill (d_ws / d_out are poisoned by the harness).
// ---------------------------------------------------------------------------
__global__ __launch_bounds__(256) void zero_kernel(float* __restrict__ p, long n) {
    long i = (long)blockIdx.x * blockDim.x + threadIdx.x;
    long stride = (long)gridDim.x * blockDim.x;
    for (; i < n; i += stride) p[i] = 0.0f;
}

// ---------------------------------------------------------------------------
// Degrees: d_v[v] += w_e[e]; d_e[e] += 1 over all incidence pairs.
// ---------------------------------------------------------------------------
__global__ __launch_bounds__(256) void degrees_kernel(
    const int* __restrict__ v_ids, const int* __restrict__ e_ids,
    const float* __restrict__ w_e, float* __restrict__ dv, float* __restrict__ de,
    int nnz) {
    int i = blockIdx.x * blockDim.x + threadIdx.x;
    if (i >= nnz) return;
    int v = v_ids[i];
    int e = e_ids[i];
    gatomic_fadd(&dv[v], w_e[e]);
    gatomic_fadd(&de[e], 1.0f);
}

// ---------------------------------------------------------------------------
// isdv[v] = d_v>0 ? rsqrt(max(d_v,1e-12)) : 0 ; ce[e] = d_e>0 ? w_e/d_e : 0
// ---------------------------------------------------------------------------
__global__ __launch_bounds__(256) void finalize_kernel(
    const float* __restrict__ dv, const float* __restrict__ de,
    const float* __restrict__ w_e, float* __restrict__ isdv,
    float* __restrict__ ce, int N, int E) {
    int i = blockIdx.x * blockDim.x + threadIdx.x;
    if (i < N) {
        float d = dv[i];
        isdv[i] = (d > 0.0f) ? rsqrtf(fmaxf(d, 1e-12f)) : 0.0f;
    }
    if (i < E) {
        float d = de[i];
        ce[i] = (d > 0.0f) ? (w_e[i] / fmaxf(d, 1e-12f)) : 0.0f;
    }
}

// ---------------------------------------------------------------------------
// WMMA GEMM: Xs = (X @ W^T + b) * isdv[row]  (f16 inputs, f32 accumulate)
// Block: 256 threads = 8 waves, M-tile = 64 rows, full C_OUT = 256 columns.
// Wave w: rows [16*(w&3), +16), columns [128*(w>>2), +128) (8 N-tiles).
// LDS: X-slab 64x32 f16 [row][k], W-slab 256x32 f16 [o][k] per 32-wide K-step.
// ---------------------------------------------------------------------------
__global__ __launch_bounds__(256) void gemm_kernel(
    const float* __restrict__ X, const float* __restrict__ W,
    const float* __restrict__ b, const float* __restrict__ isdv,
    float* __restrict__ Xs, int nRows) {
    __shared__ _Float16 Xl[64 * 32];
    __shared__ _Float16 Wl[256 * 32];

    const int tid  = threadIdx.x;
    const int lane = tid & 31;
    const int wave = tid >> 5;
    const int rowBase = blockIdx.x * 64;
    const int m0 = (wave & 3) * 16;   // wave's M sub-tile offset
    const int n0 = (wave >> 2) * 128; // wave's N half (0 or 128)

    v8f acc[8];
#pragma unroll
    for (int i = 0; i < 8; i++)
#pragma unroll
        for (int j = 0; j < 8; j++) acc[i][j] = 0.0f;

    for (int ks = 0; ks < 8; ks++) {
        const int k0 = ks * 32;
        __syncthreads();
        // X slab: 64x32 -> f16 LDS (coalesced: 32 consecutive lanes = one row)
#pragma unroll
        for (int j = 0; j < 8; j++) {
            int e = tid + 256 * j;
            int r = e >> 5, c = e & 31;
            int gr = rowBase + r;
            if (gr >= nRows) gr = nRows - 1;  // clamp; stores are masked later
            Xl[e] = (_Float16)X[(size_t)gr * 256 + k0 + c];
        }
        // W slab: 256x32 [o][k] -> f16 LDS (this *is* B = W^T column tile)
#pragma unroll
        for (int j = 0; j < 32; j++) {
            int e = tid + 256 * j;
            int o = e >> 5, c = e & 31;
            Wl[e] = (_Float16)W[(size_t)o * 256 + k0 + c];
        }
        __syncthreads();

        // A fragment (16x32 f16): lanes 0-15 row M=lane, K 0-15;
        //                          lanes 16-31 row M=lane-16, K 16-31.
        const int ar  = m0 + (lane & 15);
        const int akb = (lane < 16) ? 0 : 16;
        v16h afrag;
#pragma unroll
        for (int t = 0; t < 8; t++) {
            afrag[t]     = Xl[ar * 32 + akb + t];
            afrag[t + 8] = Xl[ar * 32 + akb + 8 + t];
        }

        // B fragments (32x16 f16): lanes 0-15 N=lane, K 0-15;
        //                          lanes 16-31 N=lane-16, K 16-31.
        const int bkb = (lane < 16) ? 0 : 16;
#pragma unroll
        for (int nt = 0; nt < 8; nt++) {
            const int o = n0 + nt * 16 + (lane & 15);
            v16h bfrag;
#pragma unroll
            for (int t = 0; t < 16; t++) bfrag[t] = Wl[o * 32 + bkb + t];
            acc[nt] = __builtin_amdgcn_wmma_f32_16x16x32_f16(
                false, afrag, false, bfrag, (short)0, acc[nt], false, false);
        }
    }

    // Epilogue: C/D layout: lanes 0-15 -> M = r, lanes 16-31 -> M = 8 + r.
    const int mHi = (lane < 16) ? 0 : 8;
    float sd[8];
    int   gmv[8];
#pragma unroll
    for (int r = 0; r < 8; r++) {
        int gm = rowBase + m0 + mHi + r;
        gmv[r] = gm;
        sd[r]  = (gm < nRows) ? isdv[gm] : 0.0f;
    }
#pragma unroll
    for (int nt = 0; nt < 8; nt++) {
        const int o = n0 + nt * 16 + (lane & 15);
        const float bias = b[o];
#pragma unroll
        for (int r = 0; r < 8; r++) {
            if (gmv[r] < nRows)
                Xs[(size_t)gmv[r] * 256 + o] = (acc[nt][r] + bias) * sd[r];
        }
    }
}

// ---------------------------------------------------------------------------
// Wave-per-nnz scatter: dst[didx[i]] += sscale(sidx[i]) * src[sidx[i]]
// 256 floats/row: lane handles 2 float4 loads + 8 HW fp32 atomics.
// scatter1: src=Xs, sidx=v_ids, dst=Ye, didx=e_ids, sscale=null
// scatter2: src=Ye, sidx=e_ids, dst=out, didx=v_ids, sscale=ce
// ---------------------------------------------------------------------------
__global__ __launch_bounds__(256) void scatter_kernel(
    const float* __restrict__ src, const int* __restrict__ sidx,
    float* __restrict__ dst, const int* __restrict__ didx,
    const float* __restrict__ sscale, int nnz) {
    int gid  = blockIdx.x * blockDim.x + threadIdx.x;
    int w    = gid >> 5;
    int lane = gid & 31;
    if (w >= nnz) return;
    int s = sidx[w];
    int d = didx[w];
    float sc = (sscale != nullptr) ? sscale[s] : 1.0f;
    const float4* sr = (const float4*)(src + (size_t)s * 256);
    float* dr = dst + (size_t)d * 256;
#pragma unroll
    for (int j = 0; j < 2; j++) {
        int c4 = lane + 32 * j;
        float4 v = sr[c4];
        gatomic_fadd(dr + c4 * 4 + 0, v.x * sc);
        gatomic_fadd(dr + c4 * 4 + 1, v.y * sc);
        gatomic_fadd(dr + c4 * 4 + 2, v.z * sc);
        gatomic_fadd(dr + c4 * 4 + 3, v.w * sc);
    }
}

// ---------------------------------------------------------------------------
// out = relu(out * isdv[row])
// ---------------------------------------------------------------------------
__global__ __launch_bounds__(256) void relu_scale_kernel(
    float* __restrict__ out, const float* __restrict__ isdv, long total) {
    long i = (long)blockIdx.x * blockDim.x + threadIdx.x;
    if (i >= total) return;
    long row = i >> 8;  // C_OUT == 256
    float v = out[i] * isdv[row];
    out[i] = (v > 0.0f) ? v : 0.0f;
}

extern "C" void kernel_launch(void* const* d_in, const int* in_sizes, int n_in,
                              void* d_out, int out_size, void* d_ws, size_t ws_size,
                              hipStream_t stream) {
    const float* X    = (const float*)d_in[0];
    const float* W    = (const float*)d_in[1];
    const float* b    = (const float*)d_in[2];
    const float* w_e  = (const float*)d_in[3];
    const int*   v_id = (const int*)d_in[4];
    const int*   e_id = (const int*)d_in[5];
    float*       out  = (float*)d_out;

    const int C   = 256;
    const int N   = in_sizes[0] / C;   // 50000
    const int E   = in_sizes[3];       // 20000
    const int NNZ = in_sizes[4];       // 800000

    // Workspace layout (floats): Xs | Ye | dv | de | isdv | ce  (~73 MB)
    float* ws   = (float*)d_ws;
    float* Xs   = ws;
    float* Ye   = Xs + (size_t)N * C;
    float* dv   = Ye + (size_t)E * C;
    float* de   = dv + N;
    float* isdv = de + E;
    float* ce   = isdv + N;

    const int ZB = 2048;  // grid-stride zero fill blocks
    zero_kernel<<<ZB, 256, 0, stream>>>(dv, (long)(N + E));          // dv+de contiguous
    zero_kernel<<<ZB, 256, 0, stream>>>(Ye, (long)E * C);
    zero_kernel<<<ZB, 256, 0, stream>>>(out, (long)N * C);

    degrees_kernel<<<(NNZ + 255) / 256, 256, 0, stream>>>(v_id, e_id, w_e, dv, de, NNZ);

    int mx = (N > E) ? N : E;
    finalize_kernel<<<(mx + 255) / 256, 256, 0, stream>>>(dv, de, w_e, isdv, ce, N, E);

    gemm_kernel<<<(N + 63) / 64, 256, 0, stream>>>(X, W, b, isdv, Xs, N);

    // vertex -> edge:  Ye[e] += Xs[v]
    scatter_kernel<<<(NNZ + 7) / 8, 256, 0, stream>>>(Xs, v_id, Ye, e_id, nullptr, NNZ);
    // edge -> vertex:  out[v] += ce[e] * Ye[e]
    scatter_kernel<<<(NNZ + 7) / 8, 256, 0, stream>>>(Ye, e_id, out, v_id, ce, NNZ);

    long total = (long)N * C;
    relu_scale_kernel<<<(unsigned)((total + 255) / 256), 256, 0, stream>>>(out, isdv, total);
}
